// GAT_En_27754078666891
// MI455X (gfx1250) — matrix operand order repaired and checked by reading, not compiled, not run on hardware
//
#include <hip/hip_runtime.h>
#include <hip/hip_bf16.h>

#define NNODES 6144
#define NFEAT  512
#define NHID   256
#define NHEADS 4
#define DHEAD  64
#define NEMBED 128
#define MAXD   256
#define LRELU  0.2f

typedef __attribute__((ext_vector_type(16))) __bf16 v16bf;
typedef __attribute__((ext_vector_type(8)))  float  v8f;

union FragAB { unsigned int u[8]; v16bf v; };
union FragC  { v8f v; float f[8]; };

__device__ __forceinline__ unsigned short f2bf(float x) {
  unsigned int u = __float_as_uint(x);
  u += 0x7FFFu + ((u >> 16) & 1u);   // round-to-nearest-even
  return (unsigned short)(u >> 16);
}

// ---------------- conversion kernels ----------------
// x: 12.6MB -> vectorized float4 loads, packed bf16x2 stores
__global__ void k_cvt_x(const float4* __restrict__ x, uint2* __restrict__ xb) {
  int i = blockIdx.x * blockDim.x + threadIdx.x;
  if (i < (NNODES * NFEAT) / 4) {
    float4 v = x[i];
    uint2 o;
    o.x = (unsigned int)f2bf(v.x) | ((unsigned int)f2bf(v.y) << 16);
    o.y = (unsigned int)f2bf(v.z) | ((unsigned int)f2bf(v.w) << 16);
    xb[i] = o;
  }
}

// Wtb[h][d][f] = W[h][f][d]  (transpose so WMMA B-columns are contiguous in K)
__global__ void k_cvt_wt(const float* __restrict__ W, unsigned short* __restrict__ Wtb) {
  int i = blockIdx.x * blockDim.x + threadIdx.x;
  if (i >= NHEADS * DHEAD * NFEAT) return;
  int h = i / (DHEAD * NFEAT);
  int r = i % (DHEAD * NFEAT);
  int d = r / NFEAT;
  int f = r % NFEAT;
  Wtb[i] = f2bf(W[(h * NFEAT + f) * DHEAD + d]);
}

__global__ void k_cvt_lin(const float* __restrict__ lw, unsigned short* __restrict__ lwb) {
  int i = blockIdx.x * blockDim.x + threadIdx.x;
  if (i < NEMBED * NHID) lwb[i] = f2bf(lw[i]);
}

// ---------------- neighbor list build ----------------
// The dominant memory pass of the whole problem: one stream over 151MB of adj.
// b128 loads: 512B per wave per vmem instruction -> issue-rate safe at 23.3 TB/s.
__global__ void k_nbr(const int* __restrict__ adj, int* __restrict__ nbr, int* __restrict__ deg) {
  __shared__ int cnt;
  const int i = blockIdx.x;
  if (threadIdx.x == 0) cnt = 0;
  __syncthreads();
  const int4* arow = (const int4*)(adj + (size_t)i * NNODES);
  for (int q = threadIdx.x; q < NNODES / 4; q += blockDim.x) {
    int4 a = arow[q];
    const int j0 = q * 4;
    if (a.x > 0) { int p = atomicAdd(&cnt, 1); if (p < MAXD) nbr[i * MAXD + p] = j0 + 0; }
    if (a.y > 0) { int p = atomicAdd(&cnt, 1); if (p < MAXD) nbr[i * MAXD + p] = j0 + 1; }
    if (a.z > 0) { int p = atomicAdd(&cnt, 1); if (p < MAXD) nbr[i * MAXD + p] = j0 + 2; }
    if (a.w > 0) { int p = atomicAdd(&cnt, 1); if (p < MAXD) nbr[i * MAXD + p] = j0 + 3; }
  }
  __syncthreads();
  if (threadIdx.x == 0) deg[i] = cnt < MAXD ? cnt : MAXD;
}

// ---------------- Wh = x @ W[h]  (bf16 WMMA, f32 accum) ----------------
// one wave per (head, 16-row block); C tile = 16 x 64 (4 WMMA n-tiles)
__global__ void k_wh_gemm(const unsigned short* __restrict__ xb,
                          const unsigned short* __restrict__ Wtb,
                          float* __restrict__ Wh) {
  const int wave = (blockIdx.x * blockDim.x + threadIdx.x) >> 5;
  const int lane = threadIdx.x & 31;
  const int MB   = NNODES / 16;
  const int h    = wave / MB;
  const int mblk = wave % MB;
  const int hi   = (lane >> 4) & 1;
  const int lo   = lane & 15;
  const unsigned short* xrow = xb + (size_t)(mblk * 16 + lo) * NFEAT;

  FragC acc[4];
#pragma unroll
  for (int t = 0; t < 4; t++)
#pragma unroll
    for (int j = 0; j < 8; j++) acc[t].f[j] = 0.f;

  for (int k0 = 0; k0 < NFEAT; k0 += 32) {
    FragAB a;
    const int kA = k0 + hi * 16;
#pragma unroll
    for (int v = 0; v < 4; v++) a.u[v] = *(const unsigned int*)(xrow + kA + 2 * v);
#pragma unroll
    for (int v = 4; v < 8; v++) a.u[v] = *(const unsigned int*)(xrow + kA + 2 * v);
    const int kB = k0 + hi * 16;
#pragma unroll
    for (int t = 0; t < 4; t++) {
      const int n = t * 16 + lo;
      const unsigned short* bcol = Wtb + ((size_t)(h * DHEAD + n)) * NFEAT;
      FragAB b;
#pragma unroll
      for (int v = 0; v < 8; v++) b.u[v] = *(const unsigned int*)(bcol + kB + 2 * v);
      acc[t].v = __builtin_amdgcn_wmma_f32_16x16x32_bf16(
          false, a.v, false, b.v, (short)0, acc[t].v, false, false);
    }
  }
#pragma unroll
  for (int t = 0; t < 4; t++) {
    const int n = t * 16 + lo;
#pragma unroll
    for (int v = 0; v < 8; v++) {
      const int m = v + hi * 8;
      Wh[((size_t)h * NNODES + mblk * 16 + m) * DHEAD + n] = acc[t].f[v];
    }
  }
}

// ---------------- s/t per-node attention terms ----------------
__global__ void k_st(const float* __restrict__ Wh, const float* __restrict__ asrc,
                     const float* __restrict__ adst, float* __restrict__ sA,
                     float* __restrict__ tA) {
  int i = blockIdx.x * blockDim.x + threadIdx.x;
  if (i >= NHEADS * NNODES) return;
  int h = i / NNODES;
  const float* wh = Wh + (size_t)i * DHEAD;
  float s = 0.f, t = 0.f;
#pragma unroll 8
  for (int d = 0; d < DHEAD; d++) {
    float w = wh[d];
    s += w * asrc[h * DHEAD + d];
    t += w * adst[h * DHEAD + d];
  }
  sA[i] = s;
  tA[i] = t;
}

// ---------------- sparse masked softmax + aggregate + ELU ----------------
// 64 threads per (node, head): thread d owns output dim d
__global__ void k_agg(const float* __restrict__ Wh, const float* __restrict__ sA,
                      const float* __restrict__ tA, const int* __restrict__ nbr,
                      const int* __restrict__ deg, unsigned short* __restrict__ hb) {
  __shared__ float ebuf[MAXD];
  __shared__ int   jbuf[MAXD];
  __shared__ float red[64];
  const int i = blockIdx.x;
  const int h = blockIdx.y;
  const int tid = threadIdx.x;
  const int dg = deg[i];
  const float si = sA[h * NNODES + i];

  float me = -3.0e38f;
  for (int idx = tid; idx < dg; idx += 64) {
    int j = nbr[i * MAXD + idx];
    float e = si + tA[h * NNODES + j];
    e = e > 0.f ? e : LRELU * e;        // leaky relu
    ebuf[idx] = e;
    jbuf[idx] = j;
    me = fmaxf(me, e);
  }
  red[tid] = me;
  __syncthreads();
  for (int off = 32; off > 0; off >>= 1) {
    if (tid < off) red[tid] = fmaxf(red[tid], red[tid + off]);
    __syncthreads();
  }
  const float m = red[0];
  __syncthreads();

  float ls = 0.f;
  for (int idx = tid; idx < dg; idx += 64) {
    float p = expf(ebuf[idx] - m);
    ebuf[idx] = p;
    ls += p;
  }
  red[tid] = ls;
  __syncthreads();
  for (int off = 32; off > 0; off >>= 1) {
    if (tid < off) red[tid] += red[tid + off];
    __syncthreads();
  }
  const float ssum = red[0];

  const int d = tid;                     // 0..63
  float acc = 0.f;
  if (dg > 0) {
    for (int idx = 0; idx < dg; idx++) {
      int j = jbuf[idx];
      acc += ebuf[idx] * Wh[((size_t)h * NNODES + j) * DHEAD + d];
    }
    acc *= (1.f / ssum);
  } else {
    // all entries NEG_INF -> softmax is uniform over all N nodes (matches jax)
    for (int j = 0; j < NNODES; j++)
      acc += Wh[((size_t)h * NNODES + j) * DHEAD + d];
    acc *= (1.f / (float)NNODES);
  }
  acc = acc > 0.f ? acc : expm1f(acc);   // ELU
  hb[(size_t)i * NHID + h * DHEAD + d] = f2bf(acc);
}

// ---------------- out = elu(hcat @ lin_w^T + b)  (bf16 WMMA) ----------------
// one wave per (16-row block, 64-col half); K = 256
__global__ void k_out_gemm(const unsigned short* __restrict__ hb,
                           const unsigned short* __restrict__ lwb,
                           const float* __restrict__ bias,
                           float* __restrict__ out) {
  const int wave = (blockIdx.x * blockDim.x + threadIdx.x) >> 5;
  const int lane = threadIdx.x & 31;
  const int mblk = wave >> 1;
  const int nt2  = wave & 1;
  const int hi   = (lane >> 4) & 1;
  const int lo   = lane & 15;
  const unsigned short* arow = hb + (size_t)(mblk * 16 + lo) * NHID;

  FragC acc[4];
#pragma unroll
  for (int t = 0; t < 4; t++)
#pragma unroll
    for (int j = 0; j < 8; j++) acc[t].f[j] = 0.f;

  for (int k0 = 0; k0 < NHID; k0 += 32) {
    FragAB a;
    const int kA = k0 + hi * 16;
#pragma unroll
    for (int v = 0; v < 4; v++) a.u[v] = *(const unsigned int*)(arow + kA + 2 * v);
#pragma unroll
    for (int v = 4; v < 8; v++) a.u[v] = *(const unsigned int*)(arow + kA + 2 * v);
    const int kB = k0 + hi * 16;
#pragma unroll
    for (int t = 0; t < 4; t++) {
      const int e = nt2 * 64 + t * 16 + lo;
      const unsigned short* bcol = lwb + (size_t)e * NHID;  // lin_w[e][k], k contiguous
      FragAB b;
#pragma unroll
      for (int v = 0; v < 8; v++) b.u[v] = *(const unsigned int*)(bcol + kB + 2 * v);
      acc[t].v = __builtin_amdgcn_wmma_f32_16x16x32_bf16(
          false, a.v, false, b.v, (short)0, acc[t].v, false, false);
    }
  }
#pragma unroll
  for (int t = 0; t < 4; t++) {
    const int col = nt2 * 64 + t * 16 + lo;
    const float bi = bias[col];
#pragma unroll
    for (int v = 0; v < 8; v++) {
      const int row = mblk * 16 + v + hi * 8;
      float val = acc[t].f[v] + bi;
      val = val > 0.f ? val : expm1f(val);  // ELU
      out[(size_t)row * NEMBED + col] = val;
    }
  }
}

extern "C" void kernel_launch(void* const* d_in, const int* in_sizes, int n_in,
                              void* d_out, int out_size, void* d_ws, size_t ws_size,
                              hipStream_t stream) {
  (void)in_sizes; (void)n_in; (void)out_size; (void)ws_size;
  const float* x    = (const float*)d_in[0];
  const int*   adj  = (const int*)  d_in[1];
  const float* W    = (const float*)d_in[2];
  const float* asrc = (const float*)d_in[3];
  const float* adst = (const float*)d_in[4];
  const float* linw = (const float*)d_in[5];
  const float* linb = (const float*)d_in[6];
  float* out = (float*)d_out;

  char* ws = (char*)d_ws;
  size_t off = 0;
  auto alloc = [&](size_t bytes) -> char* {
    char* p = ws + off;
    off = (off + bytes + 255) & ~(size_t)255;
    return p;
  };
  unsigned short* xb  = (unsigned short*)alloc((size_t)NNODES * NFEAT * 2);
  unsigned short* Wtb = (unsigned short*)alloc((size_t)NHEADS * DHEAD * NFEAT * 2);
  unsigned short* lwb = (unsigned short*)alloc((size_t)NEMBED * NHID * 2);
  float* Wh = (float*)alloc((size_t)NHEADS * NNODES * DHEAD * 4);
  float* sA = (float*)alloc((size_t)NHEADS * NNODES * 4);
  float* tA = (float*)alloc((size_t)NHEADS * NNODES * 4);
  int* nbr  = (int*)alloc((size_t)NNODES * MAXD * 4);
  int* dgA  = (int*)alloc((size_t)NNODES * 4);
  unsigned short* hb = (unsigned short*)alloc((size_t)NNODES * NHID * 2);

  k_cvt_x  <<<((NNODES * NFEAT / 4) + 255) / 256, 256, 0, stream>>>((const float4*)x, (uint2*)xb);
  k_cvt_wt <<<(NHEADS * DHEAD * NFEAT + 255) / 256, 256, 0, stream>>>(W, Wtb);
  k_cvt_lin<<<(NEMBED * NHID + 255) / 256, 256, 0, stream>>>(linw, lwb);
  k_nbr    <<<NNODES, 256, 0, stream>>>(adj, nbr, dgA);
  k_wh_gemm<<<(NHEADS * (NNODES / 16)) / 8, 256, 0, stream>>>(xb, Wtb, Wh);   // 192 blocks
  k_st     <<<(NHEADS * NNODES + 255) / 256, 256, 0, stream>>>(Wh, asrc, adst, sA, tA);
  k_agg    <<<dim3(NNODES, NHEADS), 64, 0, stream>>>(Wh, sA, tA, nbr, dgA, hb);
  k_out_gemm<<<((NNODES / 16) * 2) / 8, 256, 0, stream>>>(hb, lwb, linb, out); // 96 blocks
}
